// TernaryConv2d_49065706389532
// MI455X (gfx1250) — matrix operand order repaired and checked by reading, not compiled, not run on hardware
//
#include <hip/hip_runtime.h>

// ---- CDNA5 vector types for WMMA ----
typedef __bf16 bf16_t;
typedef __bf16 bf16x4  __attribute__((ext_vector_type(4)));   // 8 B
typedef __bf16 bf16x8  __attribute__((ext_vector_type(8)));   // 16 B
typedef __bf16 bf16x16 __attribute__((ext_vector_type(16)));
typedef float  f32x8   __attribute__((ext_vector_type(8)));
typedef int    v4i     __attribute__((ext_vector_type(4)));   // 16 B

typedef __attribute__((address_space(1))) v4i glob_v4i;   // global
typedef __attribute__((address_space(3))) v4i lds_v4i;    // LDS

#define THRESH  0.05f
#define CIN     256
#define COUT    256
#define HWDIM   56
#define IMG     (HWDIM * HWDIM)       // 3136
#define KTOT    (CIN * 9)             // 2304 ; K order = (r,s,c): k = rs*256 + c
#define KC2     64                    // K depth staged per buffer (2 WMMA k-steps)
#define NK2     (KTOT / KC2)          // 36 ; kb -> rs = kb>>2, cBase = (kb&3)*64
#define BM      128                   // Cout tile / block
#define BN      128                   // pixel tile / block
#define LDS_K   72                    // padded LDS row stride (bf16): 144 B (16B-aligned rows)

// Ternary-quantized weights, K-transposed to (r,s,c) order. 1.18 MB device .bss.
__device__ bf16_t g_wq[COUT * KTOT];

__device__ __forceinline__ bf16_t quant1(float w) {
    float q = 0.0f;
    if (fabsf(w) > THRESH) q = (w > 0.0f) ? 1.0f : -1.0f;
    return (bf16_t)q;
}

// async global->LDS 16-byte copy (CDNA5: GLOBAL_LOAD_ASYNC_TO_LDS_B128, ASYNCcnt)
__device__ __forceinline__ void async_copy_b128(const bf16_t* gsrc, bf16_t* ldst) {
#if __has_builtin(__builtin_amdgcn_global_load_async_to_lds_b128)
    __builtin_amdgcn_global_load_async_to_lds_b128(
        (glob_v4i*)gsrc, (lds_v4i*)ldst, 0, 0);
#else
    unsigned lds32 = (unsigned)(unsigned long long)ldst;  // LDS aperture: addr[31:0]
    unsigned long long g64 = (unsigned long long)gsrc;
    asm volatile("global_load_async_to_lds_b128 %0, %1, off"
                 :: "v"(lds32), "v"(g64) : "memory");
#endif
}

__device__ __forceinline__ void wait_async0() {
#if __has_builtin(__builtin_amdgcn_s_wait_asynccnt)
    __builtin_amdgcn_s_wait_asynccnt(0);
#else
    asm volatile("s_wait_asynccnt 0x0" ::: "memory");
#endif
}

// 16-elem bf16 fragment from two 16B-aligned 8-elem runs (ds_load_b128 class).
__device__ __forceinline__ bf16x16 ld_frag16(const bf16_t* p, const bf16_t* q) {
    bf16x8 lo = *(const bf16x8*)p;
    bf16x8 hi = *(const bf16x8*)q;
    return __builtin_shufflevector(lo, hi, 0, 1, 2, 3, 4, 5, 6, 7,
                                           8, 9, 10, 11, 12, 13, 14, 15);
}

// Pre-pass: quantize + transpose K from (c,r,s) to (r,s,c):
//   g_wq[co*2304 + rs*256 + c] = quant(w[co*2304 + c*9 + rs])
__global__ __launch_bounds__(256)
void quantize_weights(const float* __restrict__ w) {
    const int idx = blockIdx.x * 256 + threadIdx.x;   // 0 .. 589823
    const int co = idx / KTOT;
    const int kk = idx - co * KTOT;
    const int rs = kk >> 8;
    const int c  = kk & 255;
    g_wq[idx] = quant1(w[co * KTOT + c * 9 + rs]);
}

__global__ __launch_bounds__(256)
void ternary_conv2d_wmma(const float* __restrict__ x,
                         const float* __restrict__ bias,
                         float* __restrict__ out)
{
    __shared__ bf16_t As[2][BM * LDS_K];   // 2 x 128 x 64(+pad) bf16 weights
    __shared__ bf16_t Bs[2][BN * LDS_K];   // 2 x 128 x 64(+pad) bf16 activations

    const int t    = threadIdx.x;
    const int lane = t & 31;
    const int wid  = t >> 5;             // 0..7
    const int waveM = wid >> 1;          // 0..3 : 32 Cout rows
    const int waveN = wid & 1;           // 0..1 : 64 pixels
    const int half  = lane >> 4;
    const int lan   = lane & 15;

    const int pxBase = blockIdx.x * BN;
    const int coBase = blockIdx.y * BM;

    // ---- B staging: 128 px x 64 k / 256 thr -> 32 elem/thr ----
    const int spx  = t & 127;
    const int skk  = t >> 7;             // 0..1 -> k base skk*32
    const int pxgS = pxBase + spx;
    const int sb   = pxgS / IMG;
    const int shw  = pxgS - sb * IMG;
    const int sh   = shw / HWDIM;
    const int sw   = shw - sh * HWDIM;
    const long xPixBase = (long)sb * (CIN * IMG);

    // ---- A staging (async DMA): 128 rows x 64 k = 1024 x 16B ; 4 per thread ----
    const int aCol = (t & 7) * 8;        // k element offset (16B granules)
    const int aRow0 = t >> 3;            // 0..31 ; rows aRow0 + j*32

    f32x8 acc[2][4];
    const f32x8 zero = {};
    #pragma unroll
    for (int mi = 0; mi < 2; ++mi)
        #pragma unroll
        for (int ni = 0; ni < 4; ++ni) acc[mi][ni] = zero;

    float bR[32];

    auto issue_asyncA = [&](int kb, int buf) {
        #pragma unroll
        for (int j = 0; j < 4; ++j) {
            const int row = aRow0 + j * 32;
            async_copy_b128(&g_wq[(coBase + row) * KTOT + kb * KC2 + aCol],
                            &As[buf][row * LDS_K + aCol]);
        }
    };

    auto load_B = [&](int kb) {
        const int rs    = kb >> 2;       // uniform per chunk
        const int r     = rs / 3;
        const int s     = rs - r * 3;
        const int cBase = (kb & 3) * KC2;
        const int ih = sh + r - 1;
        const int iw = sw + s - 1;
        const bool valid = ((unsigned)ih < (unsigned)HWDIM) &&
                           ((unsigned)iw < (unsigned)HWDIM);
        const float* xp = x + xPixBase + (long)(cBase + skk * 32) * IMG
                            + ih * HWDIM + iw;
        if (valid) {
            #pragma unroll
            for (int p = 0; p < 32; ++p) bR[p] = xp[(long)p * IMG];
        } else {
            #pragma unroll
            for (int p = 0; p < 32; ++p) bR[p] = 0.0f;
        }
    };

    auto store_B = [&](int buf) {
        #pragma unroll
        for (int g = 0; g < 4; ++g) {
            bf16x8 v = { (bf16_t)bR[8 * g + 0], (bf16_t)bR[8 * g + 1],
                         (bf16_t)bR[8 * g + 2], (bf16_t)bR[8 * g + 3],
                         (bf16_t)bR[8 * g + 4], (bf16_t)bR[8 * g + 5],
                         (bf16_t)bR[8 * g + 6], (bf16_t)bR[8 * g + 7] };
            *(bf16x8*)&Bs[buf][spx * LDS_K + skk * 32 + g * 8] = v;
        }
    };

    // ---------- software-pipelined main loop ----------
    issue_asyncA(0, 0);
    load_B(0);
    int buf = 0;
    for (int kb = 0; kb < NK2; ++kb) {
        store_B(buf);                          // waits this chunk's B loads
        if (kb + 1 < NK2) load_B(kb + 1);      // B loads fly across barrier+WMMA
        wait_async0();                         // A tile (this buf) resident
        __syncthreads();
        if (kb + 1 < NK2) issue_asyncA(kb + 1, buf ^ 1);  // DMA next A tile

        #pragma unroll
        for (int ks = 0; ks < 2; ++ks) {       // two 32-deep WMMA k-steps
            const int kO = ks * 32;
            bf16x16 afrag[2];
            #pragma unroll
            for (int mi = 0; mi < 2; ++mi) {
                const int m = waveM * 32 + mi * 16 + lan;
                const bf16_t* pa = &As[buf][m * LDS_K + kO + half * 8];
                afrag[mi] = ld_frag16(pa, pa + 16);
            }
            bf16x16 bfrag[4];
            #pragma unroll
            for (int ni = 0; ni < 4; ++ni) {
                const int n = waveN * 64 + ni * 16 + lan;
                const bf16_t* pb = &Bs[buf][n * LDS_K + kO + half * 8];
                bfrag[ni] = ld_frag16(pb, pb + 16);
            }
            #pragma unroll
            for (int mi = 0; mi < 2; ++mi)
                #pragma unroll
                for (int ni = 0; ni < 4; ++ni)
                    acc[mi][ni] = __builtin_amdgcn_wmma_f32_16x16x32_bf16(
                        false, afrag[mi], false, bfrag[ni],
                        (short)0, acc[mi][ni], false, false);
        }
        buf ^= 1;
    }

    // ---------- epilogue: D VGPR i -> M = i + 8*half, N = lan ----------
    #pragma unroll
    for (int ni = 0; ni < 4; ++ni) {
        const int nG  = pxBase + waveN * 64 + ni * 16 + lan;
        const int b   = nG / IMG;
        const int hw2 = nG - b * IMG;
        const int oh  = hw2 / HWDIM;
        const int ow  = hw2 - oh * HWDIM;
        #pragma unroll
        for (int mi = 0; mi < 2; ++mi) {
            const int coT = coBase + waveM * 32 + mi * 16 + 8 * half;
            #pragma unroll
            for (int i = 0; i < 8; ++i) {
                const int co = coT + i;
                out[((b * COUT + co) * HWDIM + oh) * HWDIM + ow] =
                    acc[mi][ni][i] + bias[co];
            }
        }
    }
}

extern "C" void kernel_launch(void* const* d_in, const int* in_sizes, int n_in,
                              void* d_out, int out_size, void* d_ws, size_t ws_size,
                              hipStream_t stream) {
    (void)in_sizes; (void)n_in; (void)out_size; (void)d_ws; (void)ws_size;
    const float* x    = (const float*)d_in[0];
    const float* wgt  = (const float*)d_in[1];
    const float* bias = (const float*)d_in[2];
    float*       out  = (float*)d_out;

    dim3 block(256);
    hipLaunchKernelGGL(quantize_weights, dim3((COUT * KTOT) / 256), block, 0,
                       stream, wgt);
    dim3 grid(100352 / BN, COUT / BM);   // (784, 2)
    hipLaunchKernelGGL(ternary_conv2d_wmma, grid, block, 0, stream,
                       x, bias, out);
}
